// GPTLanguageModel_65541200937357
// MI455X (gfx1250) — matrix operand order repaired and checked by reading, not compiled, not run on hardware
//
#include <hip/hip_runtime.h>
#include <cstdint>
#include <cstddef>

typedef _Float16 half_t;
typedef __attribute__((ext_vector_type(16))) _Float16 v16h;
typedef __attribute__((ext_vector_type(8)))  float    v8f;

#define DEV __device__ __forceinline__

constexpr int Cdim = 1024, Hn = 24, HSd = 42, Tseq = 128, Bsz = 8;
constexpr int NTOK = Bsz * Tseq;     // 1024 rows (B*T)
constexpr int HHS  = Hn * HSd;       // 1008
constexpr int QKVN = 3 * HHS;        // 3024
constexpr int FFd  = 4096, Vv = 32000, Lnum = 24;

union Frag { v16h v; half_t e[16]; uint4 q[2]; };

// A-fragment (16x32 f16): lane holds row M=lane&15; kBase=(lane&16)?8:0;
// halves 0..7 -> K=kBase+j ; halves 8..15 -> K=kBase+16+j   (ISA 7.12.2)
DEV v16h ldsA_frag(const half_t* rowbase, int lane) {
  const half_t* p = rowbase + ((lane & 16) ? 8 : 0);
  Frag u; u.q[0] = *(const uint4*)p; u.q[1] = *(const uint4*)(p + 16);
  return u.v;
}
// B-fragment (32x16 f16): lane holds col N=lane&15; kBase=(lane&16)?8:0;
// halves 0..7 -> K=kBase+j ; halves 8..15 -> K=kBase+16+j (column-major source)
DEV v16h ldsB_frag(const half_t* colbase, int lane) {
  const half_t* p = colbase + ((lane & 16) ? 8 : 0);
  Frag u; u.q[0] = *(const uint4*)p; u.q[1] = *(const uint4*)(p + 16);
  return u.v;
}
DEV v8f wmma32(v16h a, v16h b, v8f c) {
  return __builtin_amdgcn_wmma_f32_16x16x32_f16(false, a, false, b, (short)0, c,
                                                false, false);
}

// ---------------------------------------------------------------- embedding
__global__ __launch_bounds__(256) void embed_kernel(
    const int* __restrict__ idx, const float* __restrict__ tok,
    float* __restrict__ x) {
  int row = blockIdx.x;
  int token = idx[row];
  const float* src = tok + (size_t)token * Cdim;
  float* dst = x + (size_t)row * Cdim;
  for (int c = threadIdx.x; c < Cdim; c += 256) dst[c] = src[c];
}

// ---------------------------------------------------------------- layernorm
__global__ __launch_bounds__(256) void ln_kernel(
    const float* __restrict__ x, const float* __restrict__ sc,
    const float* __restrict__ bi, half_t* __restrict__ out) {
  __shared__ float red[256];
  int row = blockIdx.x, tid = threadIdx.x;
  const float* xr = x + (size_t)row * Cdim;
  float s = 0.f;
  for (int c = tid; c < Cdim; c += 256) s += xr[c];
  red[tid] = s; __syncthreads();
  for (int st = 128; st > 0; st >>= 1) {
    if (tid < st) red[tid] += red[tid + st];
    __syncthreads();
  }
  float mu = red[0] / (float)Cdim;
  __syncthreads();
  float v = 0.f;
  for (int c = tid; c < Cdim; c += 256) { float d = xr[c] - mu; v += d * d; }
  red[tid] = v; __syncthreads();
  for (int st = 128; st > 0; st >>= 1) {
    if (tid < st) red[tid] += red[tid + st];
    __syncthreads();
  }
  float rs = rsqrtf(red[0] / (float)Cdim + 1e-5f);
  half_t* o = out + (size_t)row * Cdim;
  for (int c = tid; c < Cdim; c += 256)
    o[c] = (half_t)((xr[c] - mu) * rs * sc[c] + bi[c]);
}

// ------------------------------------------------- pack (H,C,HS) -> [C,3024]
__global__ __launch_bounds__(256) void pack_qkv_kernel(
    const float* __restrict__ Wq, const float* __restrict__ Wk,
    const float* __restrict__ Wv, half_t* __restrict__ out) {
  int id = blockIdx.x * 256 + threadIdx.x;
  if (id >= Cdim * HHS) return;
  int c = id / HHS, j = id % HHS;
  int h = j / HSd, d = j % HSd;
  size_t wi = ((size_t)h * Cdim + c) * HSd + d;
  size_t oi = (size_t)c * QKVN + j;
  out[oi]           = (half_t)Wq[wi];
  out[oi + HHS]     = (half_t)Wk[wi];
  out[oi + 2 * HHS] = (half_t)Wv[wi];
}

// --------------------------------------------------------------------- RoPE
__global__ __launch_bounds__(256) void rope_kernel(half_t* __restrict__ qkv) {
  int id = blockIdx.x * 256 + threadIdx.x;
  const int total = NTOK * Hn * (HSd / 2);
  if (id >= total) return;
  int p = id % (HSd / 2);
  int h = (id / (HSd / 2)) % Hn;
  int row = id / ((HSd / 2) * Hn);
  int t = row & (Tseq - 1);
  float freq = __expf((float)(2 * p) * (-9.210340371976184f / (float)HSd));
  float ang = (float)t * freq;
  float c = __cosf(ang), s = __sinf(ang);
  size_t base = (size_t)row * QKVN + (size_t)h * HSd + 2 * p;
  float qa = (float)qkv[base], qb2 = (float)qkv[base + 1];
  qkv[base]     = (half_t)(qa * c - qb2 * s);
  qkv[base + 1] = (half_t)(qa * s + qb2 * c);
  float ka = (float)qkv[base + HHS], kb2 = (float)qkv[base + HHS + 1];
  qkv[base + HHS]     = (half_t)(ka * c - kb2 * s);
  qkv[base + HHS + 1] = (half_t)(ka * s + kb2 * c);
}

// ---------------------------------------------------- generic WMMA GEMM
// C[M,N] = A[M,K](f16) * B[K,N](TB) (+epilogue), software-pipelined with
// double-buffered LDS: global loads for step k+1 are issued before the WMMA
// block of step k; the loadcnt wait lands at the next LDS store.
// EPI: 0 = store f16; 1 = bias+relu -> f16; 2 = f32 C += acc+bias; 3 = acc+bias -> f32
template <typename TB, int EPI>
__global__ __launch_bounds__(256) void gemm_kernel(
    const half_t* __restrict__ A, int lda, const TB* __restrict__ B,
    const float* __restrict__ bias, float* __restrict__ Cf,
    half_t* __restrict__ Ch, int M, int N, int K) {
  __shared__ __align__(16) half_t As[2][128][40];   // [m][k], pad stride
  __shared__ __align__(16) half_t BsT[2][128][40];  // [n][k] transposed
  const int tid = threadIdx.x, lane = tid & 31, wave = tid >> 5;
  const int mBase = blockIdx.y * 128, nBase = blockIdx.x * 128;
  const int wm = wave >> 2, wn = wave & 3;  // 2x4 wave grid, 64x32 per wave

  const int rA = tid >> 1, cA = (tid & 1) * 16;  // A staging: 16 halves/thread
  const int kB = tid >> 3, nB = (tid & 7) * 16;  // B staging: 16 elems/thread

  Frag aReg;            // pending A tile halves for this thread
  half_t bReg[16];      // pending B tile halves (already converted)

  auto load_tile = [&](int k0) {
    // ---- A: [mBase+rA][k0+cA .. +15] ----
    const half_t* srcA = A + (size_t)(mBase + rA) * lda + k0 + cA;
    if (k0 + cA + 16 <= K) {
      aReg.q[0] = *(const uint4*)(srcA);
      aReg.q[1] = *(const uint4*)(srcA + 8);
    } else {
      for (int i = 0; i < 16; i++) {
        int k = k0 + cA + i;
        aReg.e[i] = (k < K) ? srcA[i] : (half_t)0;
      }
    }
    // ---- B: row k0+kB, cols nBase+nB .. +15 ----
    int kk = k0 + kB;
    if (nBase + nB + 16 <= N && kk < K) {
      if constexpr (sizeof(TB) == 2) {
        const half_t* src = (const half_t*)B + (size_t)kk * N + nBase + nB;
        Frag u;
        u.q[0] = *(const uint4*)src;
        u.q[1] = *(const uint4*)(src + 8);
        for (int i = 0; i < 16; i++) bReg[i] = u.e[i];
      } else {
        const float* src = (const float*)B + (size_t)kk * N + nBase + nB;
        float t[16];
        *(float4*)&t[0]  = *(const float4*)(src);
        *(float4*)&t[4]  = *(const float4*)(src + 4);
        *(float4*)&t[8]  = *(const float4*)(src + 8);
        *(float4*)&t[12] = *(const float4*)(src + 12);
        for (int i = 0; i < 16; i++) bReg[i] = (half_t)t[i];
      }
    } else {
      for (int i = 0; i < 16; i++) {
        int n = nBase + nB + i;
        float v = 0.f;
        if (n < N && kk < K) v = (float)B[(size_t)kk * N + n];
        bReg[i] = (half_t)v;
      }
    }
    // prefetch tile k0+32 toward L2/WGP$ (global_prefetch_b8)
    if (k0 + 32 < K) {
      __builtin_prefetch(A + (size_t)(mBase + rA) * lda + k0 + 32 + cA, 0, 3);
      __builtin_prefetch((const char*)B + ((size_t)(k0 + 32 + kB) * N + nBase + nB) * sizeof(TB), 0, 3);
    }
  };

  const v8f vz = {0.f, 0.f, 0.f, 0.f, 0.f, 0.f, 0.f, 0.f};
  v8f acc[4][2];
  for (int i = 0; i < 4; i++)
    for (int j = 0; j < 2; j++) acc[i][j] = vz;

  load_tile(0);
  int buf = 0;
  for (int k0 = 0; k0 < K; k0 += 32) {
    // commit pending registers to LDS buffer `buf`
    *(uint4*)&As[buf][rA][cA]     = aReg.q[0];
    *(uint4*)&As[buf][rA][cA + 8] = aReg.q[1];
    for (int i = 0; i < 16; i++) BsT[buf][nB + i][kB] = bReg[i];
    __syncthreads();
    // issue next tile's global loads before computing (latency overlap)
    if (k0 + 32 < K) load_tile(k0 + 32);
    // compute from buffer `buf`
    v16h af[4], bf[2];
    for (int i = 0; i < 4; i++)
      af[i] = ldsA_frag(&As[buf][wm * 64 + i * 16 + (lane & 15)][0], lane);
    for (int j = 0; j < 2; j++)
      bf[j] = ldsB_frag(&BsT[buf][wn * 32 + j * 16 + (lane & 15)][0], lane);
    for (int i = 0; i < 4; i++)
      for (int j = 0; j < 2; j++) acc[i][j] = wmma32(af[i], bf[j], acc[i][j]);
    buf ^= 1;
  }

  const int rOff = (lane & 16) ? 8 : 0;
  for (int i = 0; i < 4; i++)
    for (int j = 0; j < 2; j++) {
      int gn = nBase + wn * 32 + j * 16 + (lane & 15);
      if (gn >= N) continue;
      for (int r = 0; r < 8; r++) {
        int gm = mBase + wm * 64 + i * 16 + rOff + r;
        if (gm >= M) continue;
        float v = acc[i][j][r];
        if constexpr (EPI == 0) {
          Ch[(size_t)gm * N + gn] = (half_t)v;
        } else if constexpr (EPI == 1) {
          v += bias[gn];
          Ch[(size_t)gm * N + gn] = (half_t)fmaxf(v, 0.f);
        } else if constexpr (EPI == 2) {
          v += bias[gn];
          Cf[(size_t)gm * N + gn] += v;
        } else {
          v += bias[gn];
          Cf[(size_t)gm * N + gn] = v;
        }
      }
    }
}

// ------------------------------------------- fused attention, one (b,h)/block
// LDS plan (49.9 KB):
//   smem[0 .. 2*128*72)  : Q then K staging [t][d], d zero-padded 42->64,
//                          later ALIASED by P prob strips [8 waves][16][136]
//   vT[48][136]          : V transposed [d][t], d zero-padded 42->48
__global__ __launch_bounds__(256) void attn_kernel(
    const half_t* __restrict__ qkv, half_t* __restrict__ o) {
  constexpr int QKP = 72;   // q/k staging row stride (halves)
  constexpr int PP  = 136;  // P row stride (halves)
  __shared__ __align__(16) half_t smem[2 * Tseq * QKP];  // 36,864 B
  __shared__ __align__(16) half_t vT[48][PP];            // 13,056 B
  half_t (*qS)[QKP] = (half_t(*)[QKP])smem;
  half_t (*kS)[QKP] = (half_t(*)[QKP])(smem + Tseq * QKP);

  const int tid = threadIdx.x, lane = tid & 31, wave = tid >> 5;
  const int b = blockIdx.x / Hn, h = blockIdx.x % Hn;
  const half_t* qb = qkv + (size_t)b * Tseq * QKVN + (size_t)h * HSd;
  const half_t* kb = qb + HHS;
  const half_t* vb = qb + 2 * HHS;

  // stage Q,K zero-padded into LDS (single guarded pass, coalesced over d)
  for (int i = tid; i < 2 * Tseq * 64; i += 256) {
    int which = i / (Tseq * 64);
    int rem = i - which * (Tseq * 64);
    int t = rem >> 6, d = rem & 63;
    const half_t* src = which ? kb : qb;
    half_t v = (d < HSd) ? src[(size_t)t * QKVN + d] : (half_t)0;
    smem[(size_t)which * Tseq * QKP + (size_t)t * QKP + d] = v;
  }
  // stage V transposed, zero-padded
  for (int i = tid; i < 48 * Tseq; i += 256) {
    int d = i / Tseq, t = i % Tseq;
    vT[d][t] = (d < HSd) ? vb[(size_t)t * QKVN + d] : (half_t)0;
  }
  __syncthreads();

  const int mBase = wave * 16;  // 8 waves * 16 rows = T=128
  const int rOff = (lane & 16) ? 8 : 0;
  const v8f vz = {0.f, 0.f, 0.f, 0.f, 0.f, 0.f, 0.f, 0.f};

  // ---- S = Q K^T  (head dim 42 padded to 64) ----
  v8f sacc[8];
  for (int nt = 0; nt < 8; nt++) sacc[nt] = vz;
  for (int kt = 0; kt < 2; kt++) {
    int kOfs = kt * 32;
    v16h aq = ldsA_frag(&qS[mBase + (lane & 15)][kOfs], lane);
    for (int nt = 0; nt < 8; nt++) {
      v16h bk = ldsB_frag(&kS[nt * 16 + (lane & 15)][kOfs], lane);
      sacc[nt] = wmma32(aq, bk, sacc[nt]);
    }
  }
  __syncthreads();  // q/k LDS dead; region reused for P below

  // ---- scale + causal mask ----
  const float scale = 0.15430334996209191f;  // 1/sqrt(42)
  for (int nt = 0; nt < 8; nt++)
    for (int r = 0; r < 8; r++) {
      int row = mBase + rOff + r, col = nt * 16 + (lane & 15);
      float s = sacc[nt][r] * scale;
      sacc[nt][r] = (col <= row) ? s : -1e30f;
    }

  // ---- register softmax; xor-shuffles 1/2/4/8 stay inside each 16-lane half ----
  float inv[8];
  for (int r = 0; r < 8; r++) {
    float m = -1e30f;
    for (int nt = 0; nt < 8; nt++) {
      float v = sacc[nt][r];
      for (int msk = 1; msk < 16; msk <<= 1) v = fmaxf(v, __shfl_xor(v, msk, 32));
      m = fmaxf(m, v);
    }
    float s = 0.f;
    for (int nt = 0; nt < 8; nt++) {
      float e = __expf(sacc[nt][r] - m);
      sacc[nt][r] = e;
      float t = e;
      for (int msk = 1; msk < 16; msk <<= 1) t += __shfl_xor(t, msk, 32);
      s += t;
    }
    inv[r] = 1.f / s;
  }

  // ---- write P strip (aliases q/k staging region) ----
  half_t (*P)[PP] = (half_t(*)[PP])(smem + (size_t)wave * 16 * PP);
  for (int nt = 0; nt < 8; nt++)
    for (int r = 0; r < 8; r++)
      P[rOff + r][nt * 16 + (lane & 15)] = (half_t)(sacc[nt][r] * inv[r]);
  __syncthreads();

  // ---- O = P V  (M=16 strip, K=128, N=48 padded) ----
  v8f oacc[3];
  for (int nt = 0; nt < 3; nt++) oacc[nt] = vz;
  for (int kt = 0; kt < 4; kt++) {
    int kOfs = kt * 32;
    v16h ap = ldsA_frag(&P[lane & 15][kOfs], lane);
    for (int nt = 0; nt < 3; nt++) {
      int d = nt * 16 + (lane & 15);
      v16h bv = ldsB_frag(&vT[d][kOfs], lane);
      oacc[nt] = wmma32(ap, bv, oacc[nt]);
    }
  }
  for (int nt = 0; nt < 3; nt++) {
    int d = nt * 16 + (lane & 15);
    if (d >= HSd) continue;
    for (int r = 0; r < 8; r++) {
      int t = mBase + rOff + r;
      o[(size_t)(b * Tseq + t) * HHS + (size_t)h * HSd + d] = (half_t)oacc[nt][r];
    }
  }
}

// ===========================================================================
extern "C" void kernel_launch(void* const* d_in, const int* in_sizes, int n_in,
                              void* d_out, int out_size, void* d_ws,
                              size_t ws_size, hipStream_t stream) {
  (void)in_sizes; (void)n_in; (void)out_size; (void)ws_size;
  const int*   idx   = (const int*)d_in[0];
  const float* tok   = (const float*)d_in[1];
  const float* Wq    = (const float*)d_in[2];
  const float* Wk    = (const float*)d_in[3];
  const float* Wv    = (const float*)d_in[4];
  const float* Wproj = (const float*)d_in[5];
  const float* bproj = (const float*)d_in[6];
  const float* ln1s  = (const float*)d_in[7];
  const float* ln1b  = (const float*)d_in[8];
  const float* ln2s  = (const float*)d_in[9];
  const float* ln2b  = (const float*)d_in[10];
  const float* W1    = (const float*)d_in[11];
  const float* b1    = (const float*)d_in[12];
  const float* W2    = (const float*)d_in[13];
  const float* b2    = (const float*)d_in[14];
  const float* lnfs  = (const float*)d_in[15];
  const float* lnfb  = (const float*)d_in[16];
  const float* Wlm   = (const float*)d_in[17];
  const float* blm   = (const float*)d_in[18];
  float* logits = (float*)d_out;

  char* w = (char*)d_ws;
  auto alloc = [&](size_t bytes) -> char* {
    char* p = w;
    w += (bytes + 255) & ~(size_t)255;
    return p;
  };
  float*  x    = (float*)alloc((size_t)NTOK * Cdim * 4);   // residual stream
  half_t* hbuf = (half_t*)alloc((size_t)NTOK * Cdim * 2);  // LN output (GEMM A)
  half_t* wqkv = (half_t*)alloc((size_t)Cdim * QKVN * 2);  // packed qkv weights
  half_t* qkv  = (half_t*)alloc((size_t)NTOK * QKVN * 2);
  half_t* ob   = (half_t*)alloc((size_t)NTOK * HHS * 2);   // attention out
  half_t* ff   = (half_t*)alloc((size_t)NTOK * FFd * 2);   // MLP hidden

  const size_t WqS = (size_t)Hn * Cdim * HSd;   // per-layer qkv weight stride
  const size_t WpS = (size_t)HHS * Cdim;
  const size_t W1S = (size_t)Cdim * FFd;
  const size_t W2S = (size_t)FFd * Cdim;

  dim3 blk(256);
  embed_kernel<<<NTOK, blk, 0, stream>>>(idx, tok, x);

  for (int l = 0; l < Lnum; l++) {
    pack_qkv_kernel<<<(Cdim * HHS + 255) / 256, blk, 0, stream>>>(
        Wq + l * WqS, Wk + l * WqS, Wv + l * WqS, wqkv);
    ln_kernel<<<NTOK, blk, 0, stream>>>(x, ln1s + l * Cdim, ln1b + l * Cdim, hbuf);
    // qkv = h @ Wqkv  [1024,1024]x[1024,3024] -> f16
    gemm_kernel<half_t, 0><<<dim3((QKVN + 127) / 128, NTOK / 128), blk, 0, stream>>>(
        hbuf, Cdim, wqkv, nullptr, nullptr, qkv, NTOK, QKVN, Cdim);
    rope_kernel<<<(NTOK * Hn * (HSd / 2) + 255) / 256, blk, 0, stream>>>(qkv);
    attn_kernel<<<Bsz * Hn, blk, 0, stream>>>(qkv, ob);
    // x += o @ Wproj + bproj   [1024,1008]x[1008,1024]
    gemm_kernel<float, 2><<<dim3(Cdim / 128, NTOK / 128), blk, 0, stream>>>(
        ob, HHS, Wproj + l * WpS, bproj + (size_t)l * Cdim, x, nullptr,
        NTOK, Cdim, HHS);
    ln_kernel<<<NTOK, blk, 0, stream>>>(x, ln2s + l * Cdim, ln2b + l * Cdim, hbuf);
    // ff = relu(h @ W1 + b1)   [1024,1024]x[1024,4096] -> f16
    gemm_kernel<float, 1><<<dim3(FFd / 128, NTOK / 128), blk, 0, stream>>>(
        hbuf, Cdim, W1 + l * W1S, b1 + (size_t)l * FFd, nullptr, ff,
        NTOK, FFd, Cdim);
    // x += ff @ W2 + b2        [1024,4096]x[4096,1024]
    gemm_kernel<float, 2><<<dim3(Cdim / 128, NTOK / 128), blk, 0, stream>>>(
        ff, FFd, W2 + l * W2S, b2 + (size_t)l * Cdim, x, nullptr,
        NTOK, Cdim, FFd);
  }

  ln_kernel<<<NTOK, blk, 0, stream>>>(x, lnfs, lnfb, hbuf);
  // logits = h @ Wlm + blm    [1024,1024]x[1024,32000] -> f32
  gemm_kernel<float, 3><<<dim3((Vv + 127) / 128, NTOK / 128), blk, 0, stream>>>(
      hbuf, Cdim, Wlm, blm, logits, nullptr, NTOK, Vv, Cdim);
}
